// bilinear_attention_23854248362443
// MI455X (gfx1250) — matrix operand order repaired and checked by reading, not compiled, not run on hardware
//
#include <hip/hip_runtime.h>
#include <hip/hip_bf16.h>

// ---------------------------------------------------------------------------
// bilinear attention:  xw = x @ Wb^T ; S = y @ xw^T ; P = softmax(S) ; O = P @ x
// B=8, Lx=Ly=2048, D=512, fp32 in/out.  WMMA bf16 path, flash-style softmax,
// TDM (tensor_load_to_lds) double-buffered key-chunk staging.
// ---------------------------------------------------------------------------

#define BATCH 8
#define LX    2048
#define LY    2048
#define DDIM  512

typedef __attribute__((ext_vector_type(16))) __bf16    v16bf;
typedef __attribute__((ext_vector_type(8)))  float     v8f;
typedef __attribute__((ext_vector_type(4)))  unsigned  u32x4;
typedef __attribute__((ext_vector_type(4)))  unsigned  uint32x4;
typedef __attribute__((ext_vector_type(8)))  int       int32x8;
typedef __attribute__((ext_vector_type(4)))  int       int32x4;

union V16 {
    v16bf bf;
    u32x4 q[2];
};

// LDS layout for the flash kernel (dynamic shared memory starts at LDS offset 0)
#define SY_OFF    0                               // 64 x 512 bf16  = 64 KB
#define SP_OFF    (64 * DDIM * 2)                 // 8 waves x 16x32 bf16 = 8 KB
#define SXW_OFF   (SP_OFF + 8 * 16 * 32 * 2)      // xw chunk double buffer
#define SXW_BYTES (32 * DDIM * 2)                 // 32 KB per buffer
#define LDS_TOTAL (SXW_OFF + 2 * SXW_BYTES)       // 136 KB

__device__ __forceinline__ unsigned short f2bfraw(float f) {
    union { float f; unsigned u; } v; v.f = f;
    unsigned u = v.u;
    unsigned r = u + 0x7FFFu + ((u >> 16) & 1u);   // round-to-nearest-even
    return (unsigned short)(r >> 16);
}

__device__ __forceinline__ v8f wmma_bf16(v16bf a, v16bf b, v8f c) {
    // (neg_a, A, neg_b, B, c_mod, C, reuse_a, reuse_b)
    return __builtin_amdgcn_wmma_f32_16x16x32_bf16(false, a, false, b,
                                                   (short)0, c, false, false);
}

// ---------------------------------------------------------------------------
// TDM: DMA a 32-row x 512-col bf16 tile (row-major, stride 512) into LDS.
// D# built per CDNA5 ISA section 8: group0 = {flags, lds_addr, global_addr,
// type=2}; group1 = {data_size=2B, tensor_dim0/1, tile_dim0/1, dim0 stride}.
// Issued by one wave; completion tracked with TENSORcnt.
// This toolchain exposes the 6-arg builtin (clang-23 / therock-10.0 headers).
// ---------------------------------------------------------------------------
__device__ __forceinline__ void tdm_load_xw_chunk(const unsigned short* gptr,
                                                  unsigned lds_byte_off) {
    unsigned long long ga = (unsigned long long)(size_t)gptr;
    uint32x4 g0;
    g0[0] = 1u;                                   // count=1, user descriptor
    g0[1] = lds_byte_off;                         // lds_addr (bytes)
    g0[2] = (unsigned)(ga & 0xFFFFFFFFu);         // global_addr[31:0]
    g0[3] = (unsigned)(ga >> 32) | (2u << 30);    // global_addr[56:32] | type=2
    int32x8 g1;
    g1[0] = 0x00010000;                           // data_size=1 (2 bytes), no multicast
    g1[1] = (int)((unsigned)DDIM << 16);          // tensor_dim0 = 512   (bits 79:48 lo)
    g1[2] = (int)(32u << 16);                     // tensor_dim1 = 32    (bits 111:80 lo)
    g1[3] = (int)((unsigned)DDIM << 16);          // tile_dim0 = 512     (bits 127:112)
    g1[4] = 32;                                   // tile_dim1 = 32      (bits 143:128)
    g1[5] = DDIM;                                 // tensor_dim0_stride  (bits 207:160)
    g1[6] = 0;
    g1[7] = 0;
    int32x4 z4 = {};
    int32x8 z8 = {};
    __builtin_amdgcn_tensor_load_to_lds(g0, g1, z4, z4, z8, 0);
}

// ---------------------------------------------------------------------------
// Kernel 1a: fp32 x -> bf16 row-major + bf16 transposed [B, D, LX]
// ---------------------------------------------------------------------------
__global__ __launch_bounds__(256)
void bilattn_convert_x(const float* __restrict__ x,
                       unsigned short* __restrict__ xbf,
                       unsigned short* __restrict__ xT) {
    size_t total  = (size_t)BATCH * LX * DDIM;
    size_t stride = (size_t)gridDim.x * blockDim.x;
    for (size_t i = (size_t)blockIdx.x * blockDim.x + threadIdx.x; i < total; i += stride) {
        unsigned short h = f2bfraw(x[i]);
        xbf[i] = h;
        size_t d  = i % DDIM;
        size_t t  = i / DDIM;
        size_t li = t % LX;
        size_t bb = t / LX;
        xT[(bb * DDIM + d) * LX + li] = h;
    }
}

// ---------------------------------------------------------------------------
// Kernel 1b: Wb fp32 -> bf16
// ---------------------------------------------------------------------------
__global__ __launch_bounds__(256)
void bilattn_convert_w(const float* __restrict__ Wb,
                       unsigned short* __restrict__ wbf) {
    int i = blockIdx.x * blockDim.x + threadIdx.x;
    if (i < DDIM * DDIM) wbf[i] = f2bfraw(Wb[i]);
}

// ---------------------------------------------------------------------------
// Kernel 2: xw[b,i,e] = sum_d x[b,i,d] * Wb[e,d]   (bf16 in, bf16 out, f32 acc)
// ---------------------------------------------------------------------------
__global__ __launch_bounds__(256)
void bilattn_proj(const unsigned short* __restrict__ xbf,   // [B, LX, D]
                  const unsigned short* __restrict__ wbf,   // [D, D] (row e major)
                  unsigned short* __restrict__ xw) {        // [B, LX, D] bf16
    const int lane    = threadIdx.x & 31;
    const int wid     = threadIdx.x >> 5;
    const int laneRow = lane & 15;
    const int laneHi  = lane >> 4;

    const int i0 = blockIdx.x * 16;
    const int e0 = (blockIdx.y * 8 + wid) * 16;
    const int b  = blockIdx.z;

    const unsigned short* xb = xbf + (size_t)b * LX * DDIM;

    v8f C = {};
#pragma unroll 4
    for (int k = 0; k < DDIM; k += 32) {
        V16 a, bb;
        // A: x rows (contiguous K: lanes 0-15 K 0-15, lanes 16-31 K 16-31)
        const unsigned short* pa = xb + (size_t)(i0 + laneRow) * DDIM + k + laneHi * 16;
        a.q[0] = *(const u32x4*)(pa);
        a.q[1] = *(const u32x4*)(pa + 8);
        // B: Wb rows e0+n as N, contiguous K per lane
        const unsigned short* pb = wbf + (size_t)(e0 + laneRow) * DDIM + k + laneHi * 16;
        bb.bf = *(const v16bf*)pb;
        C = wmma_bf16(a.bf, bb.bf, C);
    }

    // C layout: VGPR vi of lane L -> row vi + 8*(L/16), col L%16
    unsigned short* o = xw + ((size_t)b * LX + i0) * DDIM + e0;
#pragma unroll
    for (int vi = 0; vi < 8; vi++) {
        int r = vi + 8 * laneHi;
        o[(size_t)r * DDIM + laneRow] = f2bfraw(C[vi]);
    }
}

// ---------------------------------------------------------------------------
// Kernel 3: flash attention with TDM double-buffered key-chunk staging.
// Block = 8 waves = 64 Ly rows. Wave (rTile, dHalf) owns 16 rows x 256 cols.
// ---------------------------------------------------------------------------
__global__ __launch_bounds__(256)
void bilattn_flash(const float* __restrict__ y,            // [B, LY, D] fp32
                   const unsigned short* __restrict__ xw,  // [B, LX, D] bf16
                   const unsigned short* __restrict__ xT,  // [B, D, LX] bf16
                   float* __restrict__ out) {              // [B, LY, D] fp32
    extern __shared__ char smem_raw[];
    unsigned short* sy = (unsigned short*)(smem_raw + SY_OFF);
    unsigned short* sp = (unsigned short*)(smem_raw + SP_OFF);

    const int b        = blockIdx.y;
    const int rowBlock = blockIdx.x * 64;
    const int tid      = threadIdx.x;
    const int lane     = tid & 31;
    const int wid      = tid >> 5;
    const int laneRow  = lane & 15;
    const int laneHi   = lane >> 4;

    const unsigned short* xwB = xw + (size_t)b * LX * DDIM;
    const unsigned short* xTB = xT + (size_t)b * DDIM * LX;

    // ---- prologue: DMA first key chunk while we stage y into LDS -----------
    if (wid == 0) tdm_load_xw_chunk(xwB, SXW_OFF);

    const float* yb = y + ((size_t)b * LY + rowBlock) * DDIM;
    for (int i = tid; i < 64 * DDIM; i += 256) sy[i] = f2bfraw(yb[i]);

    if (wid == 0) __builtin_amdgcn_s_wait_tensorcnt(0);
    __syncthreads();

    const int rTile  = wid >> 1;      // 0..3 -> which 16 rows
    const int dHalf  = wid & 1;       // 0..1 -> which 256 columns of D
    const int rowOff = rTile * 16;
    const int d0     = dHalf * 256;

    v8f O[16];
#pragma unroll
    for (int t = 0; t < 16; t++) O[t] = {};
    float m[8], l[8];
#pragma unroll
    for (int i = 0; i < 8; i++) { m[i] = -1.0e30f; l[i] = 0.0f; }

    unsigned short* spw = sp + wid * (16 * 32);
    int cur = 0;

    for (int j = 0; j < LX; j += 32) {
        // issue DMA of the next chunk into the other buffer (overlaps compute)
        if (wid == 0 && (j + 32) < LX)
            tdm_load_xw_chunk(xwB + (size_t)(j + 32) * DDIM,
                              SXW_OFF + (unsigned)((cur ^ 1) * SXW_BYTES));

        const unsigned short* sxw =
            (const unsigned short*)(smem_raw + SXW_OFF + cur * SXW_BYTES);

        // ---- S = y_tile @ xw_chunk^T  (both operands from LDS) -------------
        v8f S0 = {}, S1 = {};
#pragma unroll 4
        for (int k = 0; k < DDIM; k += 32) {
            V16 a, b0v, b1v;
            const unsigned short* ya = sy + (size_t)(rowOff + laneRow) * DDIM + k + laneHi * 16;
            a.q[0] = *(const u32x4*)(ya);
            a.q[1] = *(const u32x4*)(ya + 8);
            const unsigned short* p0 = sxw + (size_t)laneRow * DDIM + k + laneHi * 16;
            b0v.bf = *(const v16bf*)p0;
            b1v.bf = *(const v16bf*)(p0 + 16 * DDIM);
            S0 = wmma_bf16(a.bf, b0v.bf, S0);
            S1 = wmma_bf16(a.bf, b1v.bf, S1);
        }

        // ---- online softmax over the 32 new columns ------------------------
        float sc[8];
#pragma unroll
        for (int vi = 0; vi < 8; vi++) {
            float v = fmaxf(S0[vi], S1[vi]);
            v = fmaxf(v, __shfl_xor(v, 1, 16));
            v = fmaxf(v, __shfl_xor(v, 2, 16));
            v = fmaxf(v, __shfl_xor(v, 4, 16));
            v = fmaxf(v, __shfl_xor(v, 8, 16));
            float mn = fmaxf(m[vi], v);
            sc[vi] = __expf(m[vi] - mn);
            m[vi]  = mn;
        }
#pragma unroll
        for (int vi = 0; vi < 8; vi++) {
            float p0 = __expf(S0[vi] - m[vi]);
            float p1 = __expf(S1[vi] - m[vi]);
            S0[vi] = p0; S1[vi] = p1;
            float s = p0 + p1;
            s += __shfl_xor(s, 1, 16);
            s += __shfl_xor(s, 2, 16);
            s += __shfl_xor(s, 4, 16);
            s += __shfl_xor(s, 8, 16);
            l[vi] = l[vi] * sc[vi] + s;
        }
#pragma unroll
        for (int t = 0; t < 16; t++)
#pragma unroll
            for (int vi = 0; vi < 8; vi++) O[t][vi] *= sc[vi];

        // ---- repack P (C layout, f32) -> bf16 A layout via wave-private LDS -
#pragma unroll
        for (int vi = 0; vi < 8; vi++) {
            int r = vi + 8 * laneHi;
            spw[r * 32 + laneRow]      = f2bfraw(S0[vi]);
            spw[r * 32 + 16 + laneRow] = f2bfraw(S1[vi]);
        }
        V16 ap;
        const unsigned short* pp = spw + laneRow * 32 + laneHi * 16;
        ap.q[0] = *(const u32x4*)(pp);
        ap.q[1] = *(const u32x4*)(pp + 8);

        // ---- O += P @ x_chunk  (B operand streamed from transposed x) ------
#pragma unroll 4
        for (int t = 0; t < 16; t++) {
            V16 bx;
            const unsigned short* px =
                xTB + (size_t)(d0 + t * 16 + laneRow) * LX + j + laneHi * 16;
            bx.bf = *(const v16bf*)px;
            O[t] = wmma_bf16(ap.bf, bx.bf, O[t]);
        }

        // ---- retire the in-flight DMA, publish the next buffer -------------
        if (wid == 0) __builtin_amdgcn_s_wait_tensorcnt(0);
        __syncthreads();
        cur ^= 1;
    }

    // ---- normalize and store ------------------------------------------------
    float* outB = out + ((size_t)b * LY + rowBlock + rowOff) * DDIM + d0;
#pragma unroll
    for (int t = 0; t < 16; t++) {
#pragma unroll
        for (int vi = 0; vi < 8; vi++) {
            int r = vi + 8 * laneHi;
            outB[(size_t)r * DDIM + t * 16 + laneRow] = O[t][vi] / l[vi];
        }
    }
}

// ---------------------------------------------------------------------------
// launcher
// ---------------------------------------------------------------------------
extern "C" void kernel_launch(void* const* d_in, const int* in_sizes, int n_in,
                              void* d_out, int out_size, void* d_ws, size_t ws_size,
                              hipStream_t stream) {
    const float* x  = (const float*)d_in[0];   // [B, LX, D]
    const float* y  = (const float*)d_in[1];   // [B, LY, D]
    const float* Wb = (const float*)d_in[2];   // [D, D]
    float* out = (float*)d_out;                // [B, LY, D]

    const size_t X_BYTES = (size_t)BATCH * LX * DDIM * sizeof(unsigned short); // 16 MB
    const size_t W_BYTES = (size_t)DDIM * DDIM * sizeof(unsigned short);       // 512 KB

    char* ws = (char*)d_ws;
    unsigned short* xbf = (unsigned short*)(ws);                         // [B,LX,D] bf16
    unsigned short* xT  = (unsigned short*)(ws + X_BYTES);               // [B,D,LX] bf16
    unsigned short* wbf = (unsigned short*)(ws + 2 * X_BYTES);           // [D,D]   bf16
    unsigned short* xw  = (unsigned short*)(ws + 2 * X_BYTES + W_BYTES); // [B,LX,D] bf16

    bilattn_convert_x<<<2048, 256, 0, stream>>>(x, xbf, xT);
    bilattn_convert_w<<<(DDIM * DDIM + 255) / 256, 256, 0, stream>>>(Wb, wbf);

    // xw = x @ Wb^T
    bilattn_proj<<<dim3(LX / 16, DDIM / (16 * 8), BATCH), 256, 0, stream>>>(xbf, wbf, xw);

    // fused S/softmax/PV with TDM-staged key chunks
    bilattn_flash<<<dim3(LY / 64, BATCH), 256, LDS_TOTAL, stream>>>(y, xw, xT, out);
}
